// GraphAttentionModule_61838939128425
// MI455X (gfx1250) — matrix operand order repaired and checked by reading, not compiled, not run on hardware
//
#include <hip/hip_runtime.h>

// ---------------------------------------------------------------------------
// GAT-style module on MI455X (gfx1250). All matmuls via v_wmma_f32_16x16x32_bf16,
// K/V staging via async global->LDS (ASYNCcnt) with double buffering.
//   A[16x32] operand: lane m=lane&15 holds K=16h..16h+7 (dw0-3), 16h+8..+15 (dw4-7)
//   B[32x16] operand: lane n=lane&15 holds K=16h..16h+15 (contiguous iff B^T)
// Weights pre-transposed at convert time; P@V "V" operand from transposed LDS.
// ---------------------------------------------------------------------------

typedef __attribute__((ext_vector_type(16))) __bf16        v16bf;
typedef __attribute__((ext_vector_type(8)))  float         v8f;
typedef __attribute__((ext_vector_type(4)))  unsigned int  v4u;
typedef __attribute__((ext_vector_type(8)))  unsigned int  v8u;

#define GN      4096
#define GIN     512
#define GHEADS  8
#define GHID    64
#define GOUT    128

__device__ __forceinline__ unsigned short f2bf(float f) {
    unsigned int u = __builtin_bit_cast(unsigned int, f);
    u += 0x7FFFu + ((u >> 16) & 1u);
    return (unsigned short)(u >> 16);
}
__device__ __forceinline__ float eluf(float v) {
    return v > 0.0f ? v : (__expf(v) - 1.0f);
}
__device__ __forceinline__ v16bf op_of(v4u lo, v4u hi) {
    v8u w;
    w[0] = lo[0]; w[1] = lo[1]; w[2] = lo[2]; w[3] = lo[3];
    w[4] = hi[0]; w[5] = hi[1]; w[6] = hi[2]; w[7] = hi[3];
    return __builtin_bit_cast(v16bf, w);
}
__device__ __forceinline__ v4u ld4(const unsigned short* p) {
    return *(const v4u*)p;
}

// ---- CDNA5 async global->LDS (16B per lane), tracked by ASYNCcnt ----------
__device__ __forceinline__ unsigned int lds_off(const void* p) {
    return (unsigned int)(unsigned long long)p;   // LDS aperture: addr[31:0]
}
__device__ __forceinline__ void async_ld_b128(unsigned int lds_addr,
                                              const unsigned short* gaddr) {
    asm volatile("global_load_async_to_lds_b128 %0, %1, off"
                 :: "v"(lds_addr), "v"(gaddr) : "memory");
}
template <int Ncnt>
__device__ __forceinline__ void wait_asynccnt() {
#if __has_builtin(__builtin_amdgcn_s_wait_asynccnt)
    __builtin_amdgcn_s_wait_asynccnt(Ncnt);
#else
    asm volatile("s_wait_asynccnt %0" :: "i"(Ncnt) : "memory");
#endif
}

// ---------------------------------------------------------------------------
// elementwise f32 -> bf16
// ---------------------------------------------------------------------------
__global__ void k_f32_to_bf16(const float* __restrict__ in,
                              unsigned short* __restrict__ out, long n) {
    long i = (long)blockIdx.x * blockDim.x + threadIdx.x;
    if (i < n) out[i] = f2bf(in[i]);
}

// batched f32 [B][R][C] -> bf16 transposed [B][C][R]
__global__ void k_f32_to_bf16_t(const float* __restrict__ in,
                                unsigned short* __restrict__ out,
                                int R, int C, long n) {
    long i = (long)blockIdx.x * blockDim.x + threadIdx.x;
    if (i >= n) return;
    long rc = (long)R * C;
    long b = i / rc, rem = i - b * rc;
    int r = (int)(rem / C), c = (int)(rem - (long)r * C);
    out[b * rc + (long)c * R + r] = f2bf(in[i]);
}

// ---------------------------------------------------------------------------
// bf16 GEMM with transposed B: Out[m][n] = sum_k A[m][k]*Bt[n][k] + bias[n]
// One wave computes 16 x (NT*16). grid.x = M/16, grid.y = batch (head).
// ---------------------------------------------------------------------------
template <int KDIM, int NT>
__global__ __launch_bounds__(32)
void k_gemm_bf16(const unsigned short* __restrict__ A, int lda,
                 const unsigned short* __restrict__ Bt, long bStrideY,
                 const float* __restrict__ bias, long biasStrideY,
                 unsigned short* __restrict__ Out, int ldo, long outStrideY) {
    const int lane = threadIdx.x;
    const int half = lane >> 4;
    const int nIdx = lane & 15;
    const int rowBase = blockIdx.x * 16;
    const int y = blockIdx.y;

    const unsigned short* BtH = Bt + (long)y * bStrideY;
    const float* biasH = bias + (long)y * biasStrideY;
    unsigned short* OutH = Out + (long)y * outStrideY;

    v8f acc[NT];
#pragma unroll
    for (int t = 0; t < NT; ++t) acc[t] = (v8f){0, 0, 0, 0, 0, 0, 0, 0};

    const unsigned short* arow = A + (long)(rowBase + nIdx) * lda;
#pragma unroll 4
    for (int kb = 0; kb < KDIM; kb += 32) {
        v16bf a = op_of(ld4(arow + kb + 16 * half),
                        ld4(arow + kb + 16 * half + 8));
#pragma unroll
        for (int t = 0; t < NT; ++t) {
            const unsigned short* brow = BtH + (long)(t * 16 + nIdx) * KDIM + kb + 16 * half;
            v16bf b = op_of(ld4(brow), ld4(brow + 8));
            acc[t] = __builtin_amdgcn_wmma_f32_16x16x32_bf16(
                false, a, false, b, (short)0, acc[t], false, false);
        }
    }
#pragma unroll
    for (int t = 0; t < NT; ++t) {
        int col = t * 16 + nIdx;
        float bv = biasH[col];
#pragma unroll
        for (int r = 0; r < 8; ++r) {
            int row = rowBase + r + 8 * half;
            OutH[(long)row * ldo + col] = f2bf(acc[t][r] + bv);
        }
    }
}

// ---------------------------------------------------------------------------
// Flash attention, Q=K=V=H (bf16 [Nn][D]), adjacency mask, online softmax,
// ELU on output. 4 waves per workgroup share each K/V chunk; K/V chunks are
// double-buffered and prefetched with async global->LDS loads.
// grid.x = Nn/64, grid.y = head.
// ---------------------------------------------------------------------------
template <int D, bool OUTF32>
__global__ __launch_bounds__(128)
void k_flash_attn(const unsigned short* __restrict__ qkv, long headStride,
                  const int* __restrict__ adj, int Nn,
                  unsigned short* __restrict__ outb, float* __restrict__ outf,
                  int ldo, int colOffStride, float scale) {
    constexpr int QW = 4;          // waves (query tiles) per block
    constexpr int NT = D / 16;     // 16-col output tiles per wave
    constexpr int DV = D / 8;      // uint4s per row
    constexpr int PF = (32 * DV) / 128;  // async-load instrs per wave per chunk

    __shared__ alignas(16) unsigned short sQ[QW * 16 * D];
    __shared__ alignas(16) unsigned short sKV[2][32 * D];  // [buf][key][d]
    __shared__ alignas(16) unsigned short sKVt[D * 32];    // [d][key]
    __shared__ alignas(16) float          sS[QW][16 * 32];
    __shared__ alignas(16) unsigned short sP[QW][16 * 32];
    __shared__ float                      sRow[QW][16];

    const int tid  = threadIdx.x;
    const int w    = tid >> 5;
    const int lane = tid & 31;
    const int half = lane >> 4;
    const int nIdx = lane & 15;
    const int rowBase0 = blockIdx.x * (16 * QW);
    const int qBase = rowBase0 + w * 16;
    const int head = blockIdx.y;
    const unsigned short* H = qkv + (long)head * headStride;

    // stage the 4 query tiles (64 rows x D = 8*D uint4s), b128 copies
    for (int idx = tid; idx < 8 * D; idx += 128) {
        int row = idx / DV, d8 = (idx % DV) * 8;
        *((v4u*)&sQ[row * D + d8]) = ld4(H + (long)(rowBase0 + row) * D + d8);
    }

    // async prefetch of a 32-key chunk into sKV[buf]
    auto issue_chunk = [&](int c, int buf) {
        for (int idx = tid; idx < 4 * D; idx += 128) {     // 32*DV uint4s
            int key = idx / DV, d8 = (idx % DV) * 8;
            async_ld_b128(lds_off(&sKV[buf][key * D + d8]),
                          H + (long)(c * 32 + key) * D + d8);
        }
    };

    v8f acc[NT];
#pragma unroll
    for (int t = 0; t < NT; ++t) acc[t] = (v8f){0, 0, 0, 0, 0, 0, 0, 0};
    float m_i = -3.0e38f, l_i = 0.0f;

    const int nChunks = Nn / 32;
    issue_chunk(0, 0);

    for (int c = 0; c < nChunks; ++c) {
        const int buf = c & 1;
        // prefetch next chunk into the other buffer, then wait only for ours
        if (c + 1 < nChunks) {
            issue_chunk(c + 1, buf ^ 1);
            wait_asynccnt<PF>();
        } else {
            wait_asynccnt<0>();
        }
        __syncthreads();   // chunk c visible to all waves

        // build transposed copy (V^T) for the P@V B-operand
        for (int idx = tid; idx < 4 * D; idx += 128) {
            int key = idx / DV, d0 = (idx % DV) * 8;
            union { v4u u; unsigned short s[8]; } cc;
            cc.u = *(const v4u*)&sKV[buf][key * D + d0];
#pragma unroll
            for (int j = 0; j < 8; ++j) sKVt[(d0 + j) * 32 + key] = cc.s[j];
        }
        __syncthreads();

        // S[16x32] = Q * K^T  (two 16x16 column tiles)
        v8f s0 = (v8f){0, 0, 0, 0, 0, 0, 0, 0};
        v8f s1 = (v8f){0, 0, 0, 0, 0, 0, 0, 0};
        const unsigned short* qrow = &sQ[(w * 16 + nIdx) * D];
#pragma unroll
        for (int kk = 0; kk < D; kk += 32) {
            v16bf a = op_of(ld4(qrow + kk + 16 * half),
                            ld4(qrow + kk + 16 * half + 8));
            const unsigned short* k0 = &sKV[buf][nIdx * D + kk + 16 * half];
            const unsigned short* k1 = &sKV[buf][(16 + nIdx) * D + kk + 16 * half];
            v16bf b0 = op_of(ld4(k0), ld4(k0 + 8));
            v16bf b1 = op_of(ld4(k1), ld4(k1 + 8));
            s0 = __builtin_amdgcn_wmma_f32_16x16x32_bf16(false, a, false, b0,
                                                         (short)0, s0, false, false);
            s1 = __builtin_amdgcn_wmma_f32_16x16x32_bf16(false, a, false, b1,
                                                         (short)0, s1, false, false);
        }
#pragma unroll
        for (int r = 0; r < 8; ++r) {
            sS[w][(r + 8 * half) * 32 + nIdx] = s0[r] * scale;
            sS[w][(r + 8 * half) * 32 + 16 + nIdx] = s1[r] * scale;
        }
        __syncthreads();

        // masked online softmax; lane r (<16) owns query row r of its wave
        if (lane < 16) {
            const int qRow = qBase + lane;
            float cmax = -3.0e38f;
#pragma unroll 8
            for (int j = 0; j < 32; ++j) {
                float v = (adj[(long)qRow * Nn + c * 32 + j] > 0)
                              ? sS[w][lane * 32 + j] : -1.0e9f;
                sS[w][lane * 32 + j] = v;
                cmax = fmaxf(cmax, v);
            }
            float newm = fmaxf(m_i, cmax);
            float alpha = __expf(m_i - newm);
            float sum = 0.0f;
#pragma unroll 8
            for (int j = 0; j < 32; ++j) {
                float p = __expf(sS[w][lane * 32 + j] - newm);
                sum += p;
                sP[w][lane * 32 + j] = f2bf(p);
            }
            l_i = l_i * alpha + sum;
            m_i = newm;
            sRow[w][lane] = alpha;
        }
        __syncthreads();

        // O = O*alpha + P @ V   (V operand from sKVt)
        float al[8];
#pragma unroll
        for (int r = 0; r < 8; ++r) al[r] = sRow[w][r + 8 * half];
        const unsigned short* prow = &sP[w][nIdx * 32];
        v16bf ap = op_of(ld4(prow + 16 * half), ld4(prow + 16 * half + 8));
#pragma unroll
        for (int t = 0; t < NT; ++t) {
#pragma unroll
            for (int r = 0; r < 8; ++r) acc[t][r] *= al[r];
            const unsigned short* vcol = &sKVt[(t * 16 + nIdx) * 32 + 16 * half];
            v16bf b = op_of(ld4(vcol), ld4(vcol + 8));
            acc[t] = __builtin_amdgcn_wmma_f32_16x16x32_bf16(
                false, ap, false, b, (short)0, acc[t], false, false);
        }
        __syncthreads();
    }

    if (lane < 16) sRow[w][lane] = 1.0f / fmaxf(l_i, 1.0e-30f);
    __syncthreads();
    float inv[8];
#pragma unroll
    for (int r = 0; r < 8; ++r) inv[r] = sRow[w][r + 8 * half];

#pragma unroll
    for (int t = 0; t < NT; ++t) {
        int col = t * 16 + nIdx;
#pragma unroll
        for (int r = 0; r < 8; ++r) {
            int row = qBase + r + 8 * half;
            float v = eluf(acc[t][r] * inv[r]);
            if (OUTF32)
                outf[(long)row * ldo + col] = v;
            else
                outb[(long)row * ldo + head * colOffStride + col] = f2bf(v);
        }
    }
}

// ---------------------------------------------------------------------------
// log_softmax over rows of width 128
// ---------------------------------------------------------------------------
__global__ __launch_bounds__(128)
void k_log_softmax(const float* __restrict__ in, float* __restrict__ out) {
    __shared__ float red[128];
    const int row = blockIdx.x, c = threadIdx.x;
    float v = in[(long)row * GOUT + c];
    red[c] = v;
    __syncthreads();
    for (int s = 64; s > 0; s >>= 1) {
        if (c < s) red[c] = fmaxf(red[c], red[c + s]);
        __syncthreads();
    }
    float m = red[0];
    __syncthreads();
    red[c] = __expf(v - m);
    __syncthreads();
    for (int s = 64; s > 0; s >>= 1) {
        if (c < s) red[c] += red[c + s];
        __syncthreads();
    }
    out[(long)row * GOUT + c] = v - m - __logf(red[0]);
}

// ---------------------------------------------------------------------------
extern "C" void kernel_launch(void* const* d_in, const int* in_sizes, int n_in,
                              void* d_out, int out_size, void* d_ws, size_t ws_size,
                              hipStream_t stream) {
    (void)in_sizes; (void)n_in; (void)out_size; (void)ws_size;
    const float* x   = (const float*)d_in[0];  // [N, IN]
    const int*   adj = (const int*)  d_in[1];  // [N, N]
    const float* Wh  = (const float*)d_in[2];  // [H, IN, HID]
    const float* bh  = (const float*)d_in[3];  // [H, HID]
    const float* Wo  = (const float*)d_in[4];  // [H*HID, OUT]
    const float* bo  = (const float*)d_in[5];  // [OUT]
    float* out = (float*)d_out;                // [N, OUT]

    char* ws = (char*)d_ws;
    size_t off = 0;
    auto alloc = [&](size_t bytes) -> void* {
        void* p = ws + off;
        off = (off + bytes + 255) & ~(size_t)255;
        return p;
    };
    unsigned short* xb    = (unsigned short*)alloc((size_t)GN * GIN * 2);
    unsigned short* WhbT  = (unsigned short*)alloc((size_t)GHEADS * GIN * GHID * 2); // [H][HID][IN]
    unsigned short* WobT  = (unsigned short*)alloc((size_t)GHEADS * GHID * GOUT * 2); // [OUT][H*HID]
    unsigned short* hb    = (unsigned short*)alloc((size_t)GHEADS * GN * GHID * 2);
    unsigned short* catb  = (unsigned short*)alloc((size_t)GN * GHEADS * GHID * 2);
    unsigned short* zb    = (unsigned short*)alloc((size_t)GN * GOUT * 2);
    float*          a2f   = (float*)alloc((size_t)GN * GOUT * 4);

    // 1) downconvert; weights transposed for contiguous WMMA B-operand loads
    {
        long n1 = (long)GN * GIN;
        k_f32_to_bf16<<<(n1 + 255) / 256, 256, 0, stream>>>(x, xb, n1);
        long n2 = (long)GHEADS * GIN * GHID;
        k_f32_to_bf16_t<<<(n2 + 255) / 256, 256, 0, stream>>>(Wh, WhbT, GIN, GHID, n2);
        long n3 = (long)(GHEADS * GHID) * GOUT;
        k_f32_to_bf16_t<<<(n3 + 255) / 256, 256, 0, stream>>>(Wo, WobT, GHEADS * GHID, GOUT, n3);
    }

    // 2) h[h] = x @ Wh[h] + bh[h]  -> hb [H][N][HID]
    k_gemm_bf16<GIN, GHID / 16><<<dim3(GN / 16, GHEADS), 32, 0, stream>>>(
        xb, GIN, WhbT, (long)GIN * GHID, bh, GHID,
        hb, GHID, (long)GN * GHID);

    // 3) per-head masked flash attention + ELU -> concat layout bf16
    k_flash_attn<GHID, false><<<dim3(GN / 64, GHEADS), 128, 0, stream>>>(
        hb, (long)GN * GHID, adj, GN, catb, nullptr,
        GHEADS * GHID, GHID, 0.125f /* 1/sqrt(64) */);

    // 4) z = cat @ Wo + bo -> zb [N][OUT]
    k_gemm_bf16<GHEADS * GHID, GOUT / 16><<<dim3(GN / 16, 1), 32, 0, stream>>>(
        catb, GHEADS * GHID, WobT, 0, bo, 0,
        zb, GOUT, 0);

    // 5) second masked flash attention + ELU -> f32
    k_flash_attn<GOUT, true><<<dim3(GN / 64, 1), 128, 0, stream>>>(
        zb, 0, adj, GN, nullptr, a2f,
        GOUT, 0, 0.08838834764831845f /* 1/sqrt(128) */);

    // 6) row-wise log_softmax
    k_log_softmax<<<GN, 128, 0, stream>>>(a2f, out);
}
